// MultiheadSelfAttention_3126736191744
// MI455X (gfx1250) — matrix operand order repaired and checked by reading, not compiled, not run on hardware
//
#include <hip/hip_runtime.h>
#include <hip/hip_bf16.h>

typedef __bf16 bf16;
typedef __attribute__((ext_vector_type(16))) __bf16 v16bf;
typedef __attribute__((ext_vector_type(8)))  __bf16 v8bf;
typedef __attribute__((ext_vector_type(8)))  float  v8f;
typedef __attribute__((ext_vector_type(4)))  unsigned int u32x4;
typedef __attribute__((ext_vector_type(8)))  int i32x8;
typedef __attribute__((ext_vector_type(4)))  int i32x4;

#define D_MODEL 1024
#define NH      16
#define DK      64
#define SEQ     2048
#define BATCH   2
#define NTOK    (BATCH * SEQ)   // 4096

static __device__ __forceinline__ v8f wmma_bf16(v16bf a, v16bf b, v8f c) {
  return __builtin_amdgcn_wmma_f32_16x16x32_bf16(false, a, false, b, (short)0, c, false, false);
}

// ---- Tensor Data Mover: 2D tile (rows x rowlen bf16 elements) global -> LDS ----
// D# group0: [1:0]=count=1, [63:32]=lds_addr, [120:64]=global_addr, [127:126]=type 2
// D# group1: data_size=2B, tensor_dim0/1, tile_dim0/1, tensor_dim0_stride
static __device__ __forceinline__ void tdm_load_2d(unsigned lds_off, const void* gaddr,
                                                   unsigned rows, unsigned rowlen,
                                                   unsigned src_stride,
                                                   unsigned tdim0, unsigned tdim1) {
  unsigned long long ga = (unsigned long long)gaddr;
  u32x4 g0;
  g0[0] = 1u;                                               // count=1, user mode
  g0[1] = lds_off;                                          // lds_addr (bytes)
  g0[2] = (unsigned)(ga & 0xFFFFFFFFu);                     // global_addr[31:0]
  g0[3] = (unsigned)((ga >> 32) & 0x1FFFFFFu) | (2u << 30); // global_addr[56:32] | type=2
  i32x8 g1;
  g1[0] = (int)(1u << 16);                                  // data_size=1 (2 bytes)
  g1[1] = (int)((tdim0 & 0xFFFFu) << 16);                   // tensor_dim0[15:0]
  g1[2] = (int)((tdim0 >> 16) | ((tdim1 & 0xFFFFu) << 16)); // tensor_dim0[31:16]|tensor_dim1[15:0]
  g1[3] = (int)((tdim1 >> 16) | (rowlen << 16));            // tensor_dim1[31:16]|tile_dim0
  g1[4] = (int)(rows);                                      // tile_dim1 (tile_dim2=0)
  g1[5] = (int)(src_stride);                                // tensor_dim0_stride[31:0]
  g1[6] = 0;
  g1[7] = 0;
  i32x4 z4 = {0, 0, 0, 0};
  i32x8 z8 = {0, 0, 0, 0, 0, 0, 0, 0};
  __builtin_amdgcn_tensor_load_to_lds(g0, g1, z4, z4, z8, 0);
}

// ---- Fragment loaders per CDNA5 ISA VGPR layouts (wave32) ----

// A-matrix 16x32 bf16, row-major, leading dim ld.
static __device__ __forceinline__ v16bf load_a_frag(const bf16* __restrict__ p, int ld,
                                                    int row0, int k0, int lane) {
  int m = lane & 15, h = lane >> 4;
  const bf16* base = p + (size_t)(row0 + m) * ld + k0 + h * 8;
  v8bf lo = *reinterpret_cast<const v8bf*>(base);
  v8bf hi = *reinterpret_cast<const v8bf*>(base + 16);
  v16bf r;
#pragma unroll
  for (int i = 0; i < 8; ++i) { r[i] = lo[i]; r[i + 8] = hi[i]; }
  return r;
}

// B-matrix 32x16 bf16 where B[k][n] = Rows[n][k] (Rows row-major [N][K]).
static __device__ __forceinline__ v16bf load_b_frag(const bf16* __restrict__ p, int ld,
                                                    int col0, int k0, int lane) {
  int n = lane & 15, h = lane >> 4;
  const bf16* base = p + (size_t)(col0 + n) * ld + k0 + h * 8;
  v8bf lo = *reinterpret_cast<const v8bf*>(base);
  v8bf hi = *reinterpret_cast<const v8bf*>(base + 16);
  v16bf r;
#pragma unroll
  for (int i = 0; i < 8; ++i) { r[i] = lo[i]; r[i + 8] = hi[i]; }
  return r;
}

// A-fragment of a 16x32 bf16 tile in LDS (row-major, ld=32).
static __device__ __forceinline__ v16bf lds_a_frag(const bf16* p, int lane) {
  int m = lane & 15, h = lane >> 4;
  const bf16* base = p + m * 32 + h * 8;
  v8bf lo = *reinterpret_cast<const v8bf*>(base);
  v8bf hi = *reinterpret_cast<const v8bf*>(base + 16);
  v16bf r;
#pragma unroll
  for (int i = 0; i < 8; ++i) { r[i] = lo[i]; r[i + 8] = hi[i]; }
  return r;
}

// B-fragment from LDS tile (row-major, ld=32): column n = LDS row n.
static __device__ __forceinline__ v16bf lds_b_frag(const bf16* p, int lane) {
  int n = lane & 15, h = lane >> 4;
  const bf16* base = p + n * 32 + h * 8;
  v8bf lo = *reinterpret_cast<const v8bf*>(base);
  v8bf hi = *reinterpret_cast<const v8bf*>(base + 16);
  v16bf r;
#pragma unroll
  for (int i = 0; i < 8; ++i) { r[i] = lo[i]; r[i + 8] = hi[i]; }
  return r;
}

// ---- fp32 -> bf16 conversion ----
__global__ void cvt_f32_bf16(const float* __restrict__ in, bf16* __restrict__ out, int n) {
  int i = blockIdx.x * blockDim.x + threadIdx.x;
  if (i < n) out[i] = (bf16)in[i];
}

// ---- GEMM, TDM double-buffered LDS: Y[M][N] = A[M][K] @ W[N][K]^T ----
// Block tile 128x128, 8 waves = 4(M) x 2(N), wave tile 32x64.
template <bool OUT_F32>
__global__ void __launch_bounds__(256)
gemm_bf16_wt(const bf16* __restrict__ A, const bf16* __restrict__ W,
             bf16* __restrict__ Yb, float* __restrict__ Yf,
             int M, int N, int K) {
  __shared__ bf16 Atile[2][128 * 32];   // 8 KB per buffer
  __shared__ bf16 Btile[2][128 * 32];   // 8 KB per buffer (32 KB total)
  int tid = threadIdx.x, wave = tid >> 5, lane = tid & 31;
  int bm = blockIdx.y * 128;
  int bn = blockIdx.x * 128;
  int wm = (wave >> 1) * 32;   // 0,32,64,96
  int wn = (wave & 1) * 64;    // 0,64

  unsigned ldsA[2], ldsB[2];
  ldsA[0] = (unsigned)(unsigned long long)(uintptr_t)(void*)&Atile[0][0];
  ldsA[1] = (unsigned)(unsigned long long)(uintptr_t)(void*)&Atile[1][0];
  ldsB[0] = (unsigned)(unsigned long long)(uintptr_t)(void*)&Btile[0][0];
  ldsB[1] = (unsigned)(unsigned long long)(uintptr_t)(void*)&Btile[1][0];

  v8f acc[2][4];
#pragma unroll
  for (int i = 0; i < 2; ++i)
#pragma unroll
    for (int j = 0; j < 4; ++j) acc[i][j] = (v8f){0, 0, 0, 0, 0, 0, 0, 0};

  // prologue: stage k=0 tiles into buffer 0
  if (wave == 0) {
    tdm_load_2d(ldsA[0], A + (size_t)bm * K, 128, 32, (unsigned)K, (unsigned)K, (unsigned)M);
    tdm_load_2d(ldsB[0], W + (size_t)bn * K, 128, 32, (unsigned)K, (unsigned)K, (unsigned)N);
    __builtin_amdgcn_s_wait_tensorcnt(0);
  }
  __syncthreads();

  int cur = 0;
  for (int k0 = 0; k0 < K; k0 += 32) {
    int nxt = cur ^ 1;
    bool more = (k0 + 32) < K;
    if (wave == 0 && more) {   // overlap next DMA with current compute
      tdm_load_2d(ldsA[nxt], A + (size_t)bm * K + k0 + 32, 128, 32, (unsigned)K,
                  (unsigned)K, (unsigned)M);
      tdm_load_2d(ldsB[nxt], W + (size_t)bn * K + k0 + 32, 128, 32, (unsigned)K,
                  (unsigned)K, (unsigned)N);
    }

    v16bf a0 = lds_a_frag(&Atile[cur][wm * 32], lane);
    v16bf a1 = lds_a_frag(&Atile[cur][(wm + 16) * 32], lane);
#pragma unroll
    for (int j = 0; j < 4; ++j) {
      v16bf b = lds_b_frag(&Btile[cur][(wn + j * 16) * 32], lane);
      acc[0][j] = wmma_bf16(a0, b, acc[0][j]);
      acc[1][j] = wmma_bf16(a1, b, acc[1][j]);
    }

    __syncthreads();                                   // all reads of cur complete
    if (wave == 0 && more) __builtin_amdgcn_s_wait_tensorcnt(0);
    __syncthreads();                                   // next buffer landed
    cur = nxt;
  }

  int n = lane & 15, h = lane >> 4;
#pragma unroll
  for (int i = 0; i < 2; ++i)
#pragma unroll
    for (int j = 0; j < 4; ++j)
#pragma unroll
      for (int r = 0; r < 8; ++r) {
        int row = bm + wm + i * 16 + r + h * 8;
        int col = bn + wn + j * 16 + n;
        if (OUT_F32) Yf[(size_t)row * N + col] = acc[i][j][r];
        else         Yb[(size_t)row * N + col] = (bf16)acc[i][j][r];
      }
}

// ---- RoPE + head reshape; also writes V transposed [B,H,dk,S] ----
__global__ void __launch_bounds__(256)
rope_reshape(const bf16* __restrict__ rowQ, const bf16* __restrict__ rowK,
             const bf16* __restrict__ rowV,
             bf16* __restrict__ Q, bf16* __restrict__ Kh, bf16* __restrict__ Vt) {
  int idx = blockIdx.x * blockDim.x + threadIdx.x;   // over B*H*S*(DK/2) = 2M
  int i = idx & 31;
  int s = (idx >> 5) & (SEQ - 1);
  int h = (idx >> 16) & (NH - 1);
  int b = idx >> 20;
  if (b >= BATCH) return;

  float freq = __powf(10000.0f, -(float)(2 * i) / (float)DK);
  float ang = (float)s * freq;
  float cs = __cosf(ang), sn = __sinf(ang);

  size_t rbase = ((size_t)(b * SEQ + s)) * D_MODEL + h * DK + 2 * i;
  size_t obase = (((size_t)(b * NH + h)) * SEQ + s) * DK + 2 * i;

  float q1 = (float)rowQ[rbase], q2 = (float)rowQ[rbase + 1];
  Q[obase]     = (bf16)(q1 * cs - q2 * sn);
  Q[obase + 1] = (bf16)(q1 * sn + q2 * cs);

  float k1 = (float)rowK[rbase], k2 = (float)rowK[rbase + 1];
  Kh[obase]     = (bf16)(k1 * cs - k2 * sn);
  Kh[obase + 1] = (bf16)(k1 * sn + k2 * cs);

  float v1 = (float)rowV[rbase], v2 = (float)rowV[rbase + 1];
  size_t vtb = (((size_t)(b * NH + h)) * DK + 2 * i) * SEQ + s;
  Vt[vtb]       = (bf16)v1;
  Vt[vtb + SEQ] = (bf16)v2;
}

// ---- Flash attention: wave owns a 32-row query tile, iterates 32-key blocks ----
// K/V fragments are shared across the two 16-row sub-tiles (2x WMMA per fragment).
__global__ void __launch_bounds__(128)
attn_kernel(const bf16* __restrict__ Q, const bf16* __restrict__ K,
            const bf16* __restrict__ Vt, bf16* __restrict__ O) {
  __shared__ bf16 plds[4][32 * 32];   // per-wave P tile (32 q-rows x 32 keys)
  int tid = threadIdx.x, wave = tid >> 5, lane = tid & 31;
  int gidx = blockIdx.x * 4 + wave;            // 0 .. B*H*(S/32)-1 = 2047
  int qt = gidx & (SEQ / 32 - 1);              // 64 q-tiles
  int bh = gidx >> 6;                          // 0..31
  int q0 = qt * 32;
  int b = bh >> 4, head = bh & 15;

  const bf16* Qh  = Q  + (size_t)bh * SEQ * DK;
  const bf16* Khd = K  + (size_t)bh * SEQ * DK;
  const bf16* Vh  = Vt + (size_t)bh * DK * SEQ;

  int n = lane & 15, h = lane >> 4;

  v16bf qa[2][2];   // [row sub-tile][dk chunk]
#pragma unroll
  for (int i = 0; i < 2; ++i) {
    qa[i][0] = load_a_frag(Qh, DK, q0 + i * 16, 0, lane);
    qa[i][1] = load_a_frag(Qh, DK, q0 + i * 16, 32, lane);
  }

  float mrow[2][8], lrow[2][8];
  v8f o[2][4];
#pragma unroll
  for (int i = 0; i < 2; ++i) {
#pragma unroll
    for (int r = 0; r < 8; ++r) { mrow[i][r] = -3.0e38f; lrow[i][r] = 0.0f; }
#pragma unroll
    for (int d = 0; d < 4; ++d) o[i][d] = (v8f){0, 0, 0, 0, 0, 0, 0, 0};
  }

  const float scale = 0.125f;   // 1/sqrt(64)
  int kend = q0 + 32;
  for (int kv0 = 0; kv0 < kend; kv0 += 32) {
    v16bf kb[2][2];   // [key sub-tile][dk chunk]
#pragma unroll
    for (int jt = 0; jt < 2; ++jt) {
      kb[jt][0] = load_b_frag(Khd, DK, kv0 + jt * 16, 0, lane);
      kb[jt][1] = load_b_frag(Khd, DK, kv0 + jt * 16, 32, lane);
    }
    v8f c[2][2];   // [row sub-tile][key sub-tile]
#pragma unroll
    for (int i = 0; i < 2; ++i)
#pragma unroll
      for (int jt = 0; jt < 2; ++jt) {
        v8f cc = {0, 0, 0, 0, 0, 0, 0, 0};
        cc = wmma_bf16(qa[i][0], kb[jt][0], cc);
        cc = wmma_bf16(qa[i][1], kb[jt][1], cc);
        c[i][jt] = cc;
      }

#pragma unroll
    for (int i = 0; i < 2; ++i) {
#pragma unroll
      for (int r = 0; r < 8; ++r) {
        int qrow = q0 + i * 16 + r + h * 8;
        float s0 = c[i][0][r] * scale;
        float s1 = c[i][1][r] * scale;
        if (kv0 + n > qrow)      s0 = -3.0e38f;
        if (kv0 + 16 + n > qrow) s1 = -3.0e38f;
        float mx = fmaxf(s0, s1);
#pragma unroll
        for (int msk = 1; msk < 16; msk <<= 1) mx = fmaxf(mx, __shfl_xor(mx, msk, 32));
        float mnew = fmaxf(mrow[i][r], mx);
        float corr = __expf(mrow[i][r] - mnew);
        float p0 = __expf(s0 - mnew);
        float p1 = __expf(s1 - mnew);
        float ps = p0 + p1;
#pragma unroll
        for (int msk = 1; msk < 16; msk <<= 1) ps += __shfl_xor(ps, msk, 32);
        lrow[i][r] = lrow[i][r] * corr + ps;
        mrow[i][r] = mnew;
        o[i][0][r] *= corr; o[i][1][r] *= corr; o[i][2][r] *= corr; o[i][3][r] *= corr;
        plds[wave][(i * 16 + r + h * 8) * 32 + n]      = (bf16)p0;
        plds[wave][(i * 16 + r + h * 8) * 32 + 16 + n] = (bf16)p1;
      }
    }

    v16bf pa0 = lds_a_frag(&plds[wave][0], lane);
    v16bf pa1 = lds_a_frag(&plds[wave][16 * 32], lane);
#pragma unroll
    for (int d = 0; d < 4; ++d) {
      v16bf vb = load_b_frag(Vh, SEQ, d * 16, kv0, lane);   // shared by both sub-tiles
      o[0][d] = wmma_bf16(pa0, vb, o[0][d]);
      o[1][d] = wmma_bf16(pa1, vb, o[1][d]);
    }
  }

#pragma unroll
  for (int i = 0; i < 2; ++i)
#pragma unroll
    for (int r = 0; r < 8; ++r) {
      float inv = 1.0f / lrow[i][r];
      int row = q0 + i * 16 + r + h * 8;
      size_t base = ((size_t)(b * SEQ + row)) * D_MODEL + head * DK;
      O[base + 0  + n] = (bf16)(o[i][0][r] * inv);
      O[base + 16 + n] = (bf16)(o[i][1][r] * inv);
      O[base + 32 + n] = (bf16)(o[i][2][r] * inv);
      O[base + 48 + n] = (bf16)(o[i][3][r] * inv);
    }
}

extern "C" void kernel_launch(void* const* d_in, const int* in_sizes, int n_in,
                              void* d_out, int out_size, void* d_ws, size_t ws_size,
                              hipStream_t stream) {
  const float* x  = (const float*)d_in[0];
  const float* Wq = (const float*)d_in[1];
  const float* Wk = (const float*)d_in[2];
  const float* Wv = (const float*)d_in[3];
  const float* Wo = (const float*)d_in[4];
  float* out = (float*)d_out;

  char* ws = (char*)d_ws;
  const size_t MB = 1024 * 1024;
  bf16* xb    = (bf16*)(ws + 0 * MB);
  bf16* wqb   = (bf16*)(ws + 8 * MB);
  bf16* wkb   = (bf16*)(ws + 10 * MB);
  bf16* wvb   = (bf16*)(ws + 12 * MB);
  bf16* wob   = (bf16*)(ws + 14 * MB);
  bf16* rowQ  = (bf16*)(ws + 16 * MB);
  bf16* rowK  = (bf16*)(ws + 24 * MB);
  bf16* rowV  = (bf16*)(ws + 32 * MB);
  bf16* Qr    = (bf16*)(ws + 40 * MB);
  bf16* Kr    = (bf16*)(ws + 48 * MB);
  bf16* Vt    = (bf16*)(ws + 56 * MB);
  bf16* attnO = rowQ;

  cvt_f32_bf16<<<(NTOK * D_MODEL + 255) / 256, 256, 0, stream>>>(x, xb, NTOK * D_MODEL);
  cvt_f32_bf16<<<(D_MODEL * D_MODEL + 255) / 256, 256, 0, stream>>>(Wq, wqb, D_MODEL * D_MODEL);
  cvt_f32_bf16<<<(D_MODEL * D_MODEL + 255) / 256, 256, 0, stream>>>(Wk, wkb, D_MODEL * D_MODEL);
  cvt_f32_bf16<<<(D_MODEL * D_MODEL + 255) / 256, 256, 0, stream>>>(Wv, wvb, D_MODEL * D_MODEL);
  cvt_f32_bf16<<<(D_MODEL * D_MODEL + 255) / 256, 256, 0, stream>>>(Wo, wob, D_MODEL * D_MODEL);

  dim3 gg(D_MODEL / 128, NTOK / 128);   // (8, 32)
  gemm_bf16_wt<false><<<gg, 256, 0, stream>>>(xb, wqb, rowQ, nullptr, NTOK, D_MODEL, D_MODEL);
  gemm_bf16_wt<false><<<gg, 256, 0, stream>>>(xb, wkb, rowK, nullptr, NTOK, D_MODEL, D_MODEL);
  gemm_bf16_wt<false><<<gg, 256, 0, stream>>>(xb, wvb, rowV, nullptr, NTOK, D_MODEL, D_MODEL);

  rope_reshape<<<(BATCH * NH * SEQ * (DK / 2)) / 256, 256, 0, stream>>>(rowQ, rowK, rowV, Qr, Kr, Vt);

  attn_kernel<<<(BATCH * NH * (SEQ / 32)) / 4, 128, 0, stream>>>(Qr, Kr, Vt, attnO);

  gemm_bf16_wt<true><<<gg, 256, 0, stream>>>(attnO, wob, nullptr, out, NTOK, D_MODEL, D_MODEL);
}
